// Loss_40484361732235
// MI455X (gfx1250) — compile-verified, matrix operand order checked by reading, not run on hardware
//
#include <hip/hip_runtime.h>

typedef __attribute__((ext_vector_type(16))) _Float16 v16h;
typedef __attribute__((ext_vector_type(8)))  float    v8f;

#define N_ROWS   65536
#define DIM      512
#define THRES_F  40.0f
#define WAVES_PER_BLOCK 8
#define ROWS_PER_WAVE   16

// Squared-diff of a float4 pair into 4 consecutive f16 lanes of the A fragment.
#define SQ4(dst, idx, fv, cv)                                     \
  {                                                               \
    float d0 = (fv).x - (cv).x, d1 = (fv).y - (cv).y;             \
    float d2 = (fv).z - (cv).z, d3 = (fv).w - (cv).w;             \
    (dst)[(idx) + 0] = (_Float16)(d0 * d0);                       \
    (dst)[(idx) + 1] = (_Float16)(d1 * d1);                       \
    (dst)[(idx) + 2] = (_Float16)(d2 * d2);                       \
    (dst)[(idx) + 3] = (_Float16)(d3 * d3);                       \
  }

// Stage 1: each wave32 computes 16 rows' squared distances via
// v_wmma_f32_16x16x32_f16 with B == ones (row-sum reduction in the matrix
// pipe), applies the hinge, and the block writes one partial to d_ws.
__global__ __launch_bounds__(256) void Loss_dist_hinge_kernel(
    const float* __restrict__ features,
    const float* __restrict__ center,
    const int*   __restrict__ labels,
    float*       __restrict__ partial)
{
  __shared__ float waveSums[WAVES_PER_BLOCK];

  const int lane        = threadIdx.x & 31;
  const int waveInBlock = threadIdx.x >> 5;
  const int waveGlobal  = blockIdx.x * WAVES_PER_BLOCK + waveInBlock;
  const int rowBase     = waveGlobal * ROWS_PER_WAVE;

  // A-matrix 16x32 f16 layout (wave32):
  //   lanes 0-15  : M = lane,    K chunks {0..7, 16..23}
  //   lanes 16-31 : M = lane-16, K chunks {8..15, 24..31}
  const int half = lane >> 4;        // 0 -> low K half, 1 -> high K half
  const int m    = lane & 15;        // row within the 16-row tile
  const int row  = rowBase + m;

  const int lbl = labels[row];
  const float* fp = features + (size_t)row * DIM + half * 8;
  const float* cp = center   + (size_t)lbl * DIM + half * 8;

  v16h ones;
#pragma unroll
  for (int i = 0; i < 16; ++i) ones[i] = (_Float16)1.0f;

  v8f acc = {};  // f32 accumulator: D[m][n] = sum_k diff^2[m][k]

#pragma unroll 4
  for (int k0 = 0; k0 < DIM; k0 += 32) {
    float4 f0 = *(const float4*)(fp + k0);
    float4 f1 = *(const float4*)(fp + k0 + 4);
    float4 f2 = *(const float4*)(fp + k0 + 16);
    float4 f3 = *(const float4*)(fp + k0 + 20);
    float4 c0 = *(const float4*)(cp + k0);
    float4 c1 = *(const float4*)(cp + k0 + 4);
    float4 c2 = *(const float4*)(cp + k0 + 16);
    float4 c3 = *(const float4*)(cp + k0 + 20);

    v16h a;
    SQ4(a, 0,  f0, c0);
    SQ4(a, 4,  f1, c1);
    SQ4(a, 8,  f2, c2);
    SQ4(a, 12, f3, c3);

    // D = A x ones + C  -> every column of row m holds sum_k diff^2[m][k]
    acc = __builtin_amdgcn_wmma_f32_16x16x32_f16(
        /*neg_a=*/false, a, /*neg_b=*/false, ones,
        /*c_mod=*/(short)0, acc, /*reuse_a=*/false, /*reuse_b=*/false);
  }

  // C/D layout: VGPR r -> M=r (lanes 0-15), M=8+r (lanes 16-31), value
  // replicated across the 16 columns. Each lane sums the hinge over its
  // 8 rows; lane 0 carries rows 0-7, lane 16 carries rows 8-15.
  float psum = 0.0f;
#pragma unroll
  for (int r = 0; r < 8; ++r) {
    float dist = __builtin_sqrtf(acc[r]);
    float h    = THRES_F - dist;
    psum += fmaxf(h, 0.0f);
  }

  float lo = __shfl(psum, 0, 32);
  float hi = __shfl(psum, 16, 32);
  if (lane == 0) waveSums[waveInBlock] = lo + hi;
  __syncthreads();

  if (threadIdx.x == 0) {
    float s = 0.0f;
#pragma unroll
    for (int w = 0; w < WAVES_PER_BLOCK; ++w) s += waveSums[w];
    partial[blockIdx.x] = s;  // deterministic: fixed order, no atomics
  }
}

// Stage 2: single wave32 reduces the per-block partials deterministically
// (fixed strided accumulation + fixed shuffle tree) and writes the mean.
__global__ __launch_bounds__(32) void Loss_reduce_kernel(
    const float* __restrict__ partial,
    float*       __restrict__ out,
    int nPartial)
{
  const int lane = threadIdx.x;
  float s = 0.0f;
  for (int i = lane; i < nPartial; i += 32) s += partial[i];
#pragma unroll
  for (int off = 16; off > 0; off >>= 1) s += __shfl_down(s, off, 32);
  if (lane == 0) out[0] = s * (1.0f / (float)N_ROWS);
}

extern "C" void kernel_launch(void* const* d_in, const int* in_sizes, int n_in,
                              void* d_out, int out_size, void* d_ws, size_t ws_size,
                              hipStream_t stream) {
  (void)n_in; (void)out_size; (void)ws_size;
  const float* features = (const float*)d_in[0];   // [N, D] f32
  const float* center   = (const float*)d_in[1];   // [C, D] f32
  const int*   labels   = (const int*)d_in[2];     // [N]    i32
  float* out     = (float*)d_out;
  float* partial = (float*)d_ws;                   // nBlocks floats (2 KB)

  const int nRows   = in_sizes[2];                                 // 65536
  const int nBlocks = nRows / (ROWS_PER_WAVE * WAVES_PER_BLOCK);   // 512

  Loss_dist_hinge_kernel<<<nBlocks, 32 * WAVES_PER_BLOCK, 0, stream>>>(
      features, center, labels, partial);
  Loss_reduce_kernel<<<1, 32, 0, stream>>>(partial, out, nBlocks);
}
